// NewRobertaLayer_5540507811948
// MI455X (gfx1250) — compile-verified, matrix operand order checked by reading, not run on hardware
//
#include <hip/hip_runtime.h>
#include <cstdint>
#include <cstddef>

// ---------------- types ----------------
typedef __bf16 bf16_t;
typedef bf16_t v16bf __attribute__((ext_vector_type(16)));
typedef float  v8f   __attribute__((ext_vector_type(8)));
typedef unsigned int u32x4 __attribute__((ext_vector_type(4)));
typedef int          i32x4 __attribute__((ext_vector_type(4)));
typedef int          i32x8 __attribute__((ext_vector_type(8)));

union FragBF {
    uint32_t       d[8];
    unsigned short u[16];
    v16bf          v;
};

static __device__ inline v8f zero8() {
    v8f z;
#pragma unroll
    for (int i = 0; i < 8; ++i) z[i] = 0.0f;
    return z;
}

// fp32 -> bf16 (round to nearest even), returned as raw u16 bits
static __device__ inline unsigned short f2bfu(float f) {
    union { float f; uint32_t u; } c; c.f = f;
    uint32_t r = c.u + 0x7FFFu + ((c.u >> 16) & 1u);
    return (unsigned short)(r >> 16);
}

// model constants
#define BB   4
#define SS   1024
#define DD   1024
#define NHH  16
#define HDD  64
#define II   4096
#define BSR  4096   /* B*S rows */

// ---- CDNA5 Tensor Data Mover (TDM) support ----
#if defined(__gfx1250__) && __has_builtin(__builtin_amdgcn_tensor_load_to_lds)
#define USE_TDM 1
#endif

#ifdef USE_TDM
// Issue a TDM DMA of a 2D bf16 tile [tileH rows x tileW elems] from global
// (row pitch strideW elems) into LDS at lds_off (row-major, packed).
// D# per CDNA5 ISA 8.3/8.4: group0 = {count, lds_addr, global_addr, type=2},
// group1 = {data_size, tensor dims, tile dims, dim0 stride}.
static __device__ inline void tdm_load_2d(unsigned lds_off, const void* gaddr,
                                          unsigned tileW, unsigned tileH,
                                          unsigned tensorW, unsigned tensorH,
                                          unsigned strideW) {
    unsigned long long ga = (unsigned long long)gaddr;
    u32x4 g0;
    g0[0] = 1u;                                   // count=1 (valid), user mode
    g0[1] = lds_off;                              // lds_addr (bytes)
    g0[2] = (unsigned)ga;                         // global_addr[31:0]
    g0[3] = (unsigned)(ga >> 32) | 0x80000000u;   // global_addr[56:32] | type=2
    i32x8 g1;
    g1[0] = (int)(1u << 16);                      // workgroup_mask=0, data_size=1 (2B)
    g1[1] = (int)((tensorW & 0xFFFFu) << 16);     // tensor_dim0[15:0] @ bit48
    g1[2] = (int)(((tensorW >> 16) & 0xFFFFu) |
                  ((tensorH & 0xFFFFu) << 16));   // tensor_dim0 hi | tensor_dim1 lo
    g1[3] = (int)(((tensorH >> 16) & 0xFFFFu) |
                  (tileW << 16));                 // tensor_dim1 hi | tile_dim0
    g1[4] = (int)(tileH & 0xFFFFu);               // tile_dim1 | tile_dim2=0
    g1[5] = (int)strideW;                         // tensor_dim0_stride[31:0]
    g1[6] = 0;                                    // stride hi | tensor_dim1_stride lo
    g1[7] = 0;
    i32x4 z4 = {0, 0, 0, 0};
    i32x8 z8 = {0, 0, 0, 0, 0, 0, 0, 0};
    // 6-arg toolchain form: (g0, g1, g2, g3, g4, cpol)
    __builtin_amdgcn_tensor_load_to_lds(g0, g1, z4, z4, z8, 0);
}
#endif

// ---------------- fp32 -> bf16 convert ----------------
__global__ __launch_bounds__(256) void cvt_f32_bf16(const float* __restrict__ in,
                                                    unsigned short* __restrict__ out,
                                                    int n) {
    for (int i = blockIdx.x * blockDim.x + threadIdx.x; i < n;
         i += gridDim.x * blockDim.x)
        out[i] = f2bfu(in[i]);
}

// ---------------- tiled bf16 WMMA GEMM, TDM double-buffered ----------------
// C[M,N] = A[M,K] * B[K,N]  (A,B bf16 row-major, acc f32)
// block tile 128x128, 8 waves (2x4), wave tile 64x32 (4x2 wmma tiles), BK=32
// MODE 0: outF[row*N+col] = acc + bias[col] + res[row*N+col]
// MODE 1: QKV scatter: outB[((b*NH+head)*S + s)*HD + hd] = bf16(acc + bias)
// MODE 2: outB[row*N+col] = bf16(gelu(acc + bias))
template <int MODE>
__global__ __launch_bounds__(256) void gemm_bf16(const unsigned short* __restrict__ A,
                                                 const unsigned short* __restrict__ Bw,
                                                 const float* __restrict__ bias,
                                                 float* __restrict__ outF,
                                                 unsigned short* __restrict__ outB,
                                                 const float* __restrict__ res,
                                                 int M, int N, int K) {
    __shared__ unsigned short As[2][128 * 32];
    __shared__ unsigned short Bs[2][128 * 32];

    const int tid   = threadIdx.x;
    const int lane  = tid & 31;
    const int wid   = tid >> 5;
    const int n16   = lane & 15;
    const int hi    = lane >> 4;
    const int waveM = wid >> 2;   // 0..1
    const int waveN = wid & 3;    // 0..3
    const int rowBase = blockIdx.y * 128;
    const int colBase = blockIdx.x * 128;

    v8f acc[4][2];
#pragma unroll
    for (int i = 0; i < 4; ++i)
#pragma unroll
        for (int j = 0; j < 2; ++j) acc[i][j] = zero8();

    // stage tile (A:128x32, B:32x128) for K-step k0 into buffer `buf`
    auto stage = [&](int buf, int k0) {
#ifdef USE_TDM
        if (wid == 0) {
            tdm_load_2d((unsigned)(size_t)&As[buf][0],
                        A + (size_t)rowBase * K + k0,
                        /*tileW=*/32, /*tileH=*/128,
                        /*tensorW=*/(unsigned)K, /*tensorH=*/(unsigned)M,
                        /*strideW=*/(unsigned)K);
            tdm_load_2d((unsigned)(size_t)&Bs[buf][0],
                        Bw + (size_t)k0 * N + colBase,
                        /*tileW=*/128, /*tileH=*/32,
                        /*tensorW=*/(unsigned)N, /*tensorH=*/(unsigned)K,
                        /*strideW=*/(unsigned)N);
        }
#else
#pragma unroll
        for (int i = 0; i < 2; ++i) {
            int ci = tid + i * 256;              // 0..511
            int ra = ci >> 2;                    // A row, 4 chunks/row
            int ca = (ci & 3) << 3;
            *(uint4*)&As[buf][ra * 32 + ca] =
                *(const uint4*)&A[(size_t)(rowBase + ra) * K + k0 + ca];
            int rb = ci >> 4;                    // B row, 16 chunks/row
            int cb = (ci & 15) << 3;
            *(uint4*)&Bs[buf][rb * 128 + cb] =
                *(const uint4*)&Bw[(size_t)(k0 + rb) * N + colBase + cb];
        }
#endif
    };

    stage(0, 0);
    int cur = 0;
    for (int k0 = 0; k0 < K; k0 += 32) {
#ifdef USE_TDM
        if (wid == 0) __builtin_amdgcn_s_wait_tensorcnt((short)0);
#endif
        __syncthreads();                     // tile `cur` resident & visible
        if (k0 + 32 < K) stage(cur ^ 1, k0 + 32);   // overlap next DMA w/ compute

        const unsigned short* as = &As[cur][0];
        const unsigned short* bs = &Bs[cur][0];

        FragBF af[4];
#pragma unroll
        for (int mt = 0; mt < 4; ++mt) {
            int m = waveM * 64 + mt * 16 + n16;
#pragma unroll
            for (int v = 0; v < 8; ++v) {
                int k = ((v >> 2) << 4) + (hi << 3) + ((v & 3) << 1);
                af[mt].d[v] = *(const uint32_t*)&as[m * 32 + k];
            }
        }
        FragBF bf[2];
#pragma unroll
        for (int nt = 0; nt < 2; ++nt) {
            int n = waveN * 32 + nt * 16 + n16;
#pragma unroll
            for (int v = 0; v < 8; ++v) {
                int k = ((v >> 2) << 4) + (hi << 3) + ((v & 3) << 1);
                bf[nt].u[2 * v]     = bs[k * 128 + n];
                bf[nt].u[2 * v + 1] = bs[(k + 1) * 128 + n];
            }
        }
#pragma unroll
        for (int mt = 0; mt < 4; ++mt)
#pragma unroll
            for (int nt = 0; nt < 2; ++nt)
                acc[mt][nt] = __builtin_amdgcn_wmma_f32_16x16x32_bf16(
                    false, af[mt].v, false, bf[nt].v, (short)0, acc[mt][nt],
                    false, false);
        __syncthreads();                     // all waves done reading `cur`
        cur ^= 1;
    }

    // epilogue
#pragma unroll
    for (int mt = 0; mt < 4; ++mt) {
#pragma unroll
        for (int nt = 0; nt < 2; ++nt) {
#pragma unroll
            for (int r = 0; r < 8; ++r) {
                int grow = rowBase + waveM * 64 + mt * 16 + r + 8 * hi;
                int gcol = colBase + waveN * 32 + nt * 16 + n16;
                float v = acc[mt][nt][r] + bias[gcol];
                if (MODE == 0) {
                    outF[(size_t)grow * N + gcol] = v + res[(size_t)grow * N + gcol];
                } else if (MODE == 1) {
                    int b = grow >> 10, s = grow & 1023;
                    int head = gcol >> 6, hd = gcol & 63;
                    outB[(((size_t)(b * NHH + head)) * SS + s) * HDD + hd] = f2bfu(v);
                } else {
                    float g = 0.5f * v * (1.0f + erff(v * 0.70710678118654752f));
                    outB[(size_t)grow * N + gcol] = f2bfu(g);
                }
            }
        }
    }
}

// ---------------- attention: one wave per (b, h, 16-row query tile) ----------
// dyn LDS: float sS[16][1024] (64KB) + ushort sP[16][1024] (32KB) = 96KB
__global__ __launch_bounds__(32) void attn_kernel(const unsigned short* __restrict__ Q,
                                                  const unsigned short* __restrict__ Kb,
                                                  const unsigned short* __restrict__ Vb,
                                                  const float* __restrict__ mask,
                                                  unsigned short* __restrict__ ctxB) {
    extern __shared__ char smem[];
    float*          sS = (float*)smem;
    unsigned short* sP = (unsigned short*)(smem + 16 * 1024 * sizeof(float));

    const int lane = threadIdx.x;
    const int n16  = lane & 15;
    const int hi   = lane >> 4;
    const int bh   = blockIdx.x >> 6;     // (b*NH + h)
    const int qt   = blockIdx.x & 63;
    const int b    = bh >> 4;
    const int h    = bh & 15;
    const size_t headOff = (size_t)bh * SS * HDD;

    // Q tile (16x64) as two 16x32 A-fragments, loaded straight in A layout
    FragBF qa[2];
    const int qrow = qt * 16 + n16;
#pragma unroll
    for (int c = 0; c < 2; ++c)
#pragma unroll
        for (int v = 0; v < 8; ++v) {
            int k = 32 * c + ((v >> 2) << 4) + (hi << 3) + ((v & 3) << 1);
            qa[c].d[v] = *(const uint32_t*)&Q[headOff + (size_t)qrow * HDD + k];
        }

    // scores: S_tile = (Q . K^T)/8 + mask  -> LDS f32 strip [16][1024]
    for (int kt = 0; kt < 64; ++kt) {
        v8f s = zero8();
        const size_t krow = headOff + (size_t)(kt * 16 + n16) * HDD;
#pragma unroll
        for (int c = 0; c < 2; ++c) {
            FragBF kf;
#pragma unroll
            for (int v = 0; v < 8; ++v) {
                int k = 32 * c + ((v >> 2) << 4) + (hi << 3) + ((v & 3) << 1);
                kf.d[v] = *(const uint32_t*)&Kb[krow + k];
            }
            s = __builtin_amdgcn_wmma_f32_16x16x32_bf16(false, qa[c].v, false,
                                                        kf.v, (short)0, s,
                                                        false, false);
        }
        float mv = mask[(size_t)b * SS + kt * 16 + n16];
#pragma unroll
        for (int r = 0; r < 8; ++r)
            sS[(r + 8 * hi) * 1024 + kt * 16 + n16] = s[r] * 0.125f + mv;
    }
    __syncthreads();

    // softmax over each of the 16 rows; lane l owns row l%16, half l/16
    {
        const int row = n16;
        const int cb  = hi * 512;
        float mx = -3.4e38f;
        for (int c = 0; c < 512; ++c)
            mx = fmaxf(mx, sS[row * 1024 + cb + c]);
        mx = fmaxf(mx, __shfl_xor(mx, 16, 32));
        float sum = 0.0f;
        for (int c = 0; c < 512; ++c) {
            float e = __expf(sS[row * 1024 + cb + c] - mx);
            sS[row * 1024 + cb + c] = e;
            sum += e;
        }
        sum += __shfl_xor(sum, 16, 32);
        float inv = 1.0f / sum;
        for (int c = 0; c < 512; ++c)
            sP[row * 1024 + cb + c] = f2bfu(sS[row * 1024 + cb + c] * inv);
    }
    __syncthreads();

    // ctx = P(16x1024) . V(1024x64)
    v8f ctx[4];
#pragma unroll
    for (int nt = 0; nt < 4; ++nt) ctx[nt] = zero8();
    for (int kc = 0; kc < 32; ++kc) {
        FragBF pf;
#pragma unroll
        for (int v = 0; v < 8; ++v) {
            int k = kc * 32 + ((v >> 2) << 4) + (hi << 3) + ((v & 3) << 1);
            pf.d[v] = *(const uint32_t*)&sP[n16 * 1024 + k];
        }
#pragma unroll
        for (int nt = 0; nt < 4; ++nt) {
            FragBF vf;
#pragma unroll
            for (int v = 0; v < 8; ++v) {
                int k = kc * 32 + ((v >> 2) << 4) + (hi << 3) + ((v & 3) << 1);
                vf.u[2 * v]     = Vb[headOff + (size_t)k * HDD + nt * 16 + n16];
                vf.u[2 * v + 1] = Vb[headOff + (size_t)(k + 1) * HDD + nt * 16 + n16];
            }
            ctx[nt] = __builtin_amdgcn_wmma_f32_16x16x32_bf16(false, pf.v, false,
                                                              vf.v, (short)0,
                                                              ctx[nt], false, false);
        }
    }
    // store ctx as bf16 in [B,S,D] layout
#pragma unroll
    for (int nt = 0; nt < 4; ++nt)
#pragma unroll
        for (int r = 0; r < 8; ++r) {
            int srow = qt * 16 + r + 8 * hi;
            int dcol = h * 64 + nt * 16 + n16;
            ctxB[((size_t)b * SS + srow) * DD + dcol] = f2bfu(ctx[nt][r]);
        }
}

// ---------------- layernorm: one block per row (D=1024) ----------------
__global__ __launch_bounds__(256) void ln_kernel(const float* __restrict__ x,
                                                 const float* __restrict__ g,
                                                 const float* __restrict__ bta,
                                                 float* __restrict__ outF,
                                                 unsigned short* __restrict__ outB) {
    __shared__ float red[256];
    const int row = blockIdx.x;
    const int tid = threadIdx.x;
    const float* xr = x + (size_t)row * DD;

    float s = 0.0f;
    for (int i = tid; i < DD; i += 256) s += xr[i];
    red[tid] = s; __syncthreads();
    for (int off = 128; off > 0; off >>= 1) {
        if (tid < off) red[tid] += red[tid + off];
        __syncthreads();
    }
    float mu = red[0] * (1.0f / DD);
    __syncthreads();

    float v = 0.0f;
    for (int i = tid; i < DD; i += 256) {
        float d = xr[i] - mu; v += d * d;
    }
    red[tid] = v; __syncthreads();
    for (int off = 128; off > 0; off >>= 1) {
        if (tid < off) red[tid] += red[tid + off];
        __syncthreads();
    }
    float inv = rsqrtf(red[0] * (1.0f / DD) + 1e-12f);

    for (int i = tid; i < DD; i += 256) {
        float o = (xr[i] - mu) * inv * g[i] + bta[i];
        outF[(size_t)row * DD + i] = o;
        if (outB) outB[(size_t)row * DD + i] = f2bfu(o);
    }
}

// ---------------- launcher ----------------
extern "C" void kernel_launch(void* const* d_in, const int* in_sizes, int n_in,
                              void* d_out, int out_size, void* d_ws, size_t ws_size,
                              hipStream_t stream) {
    const float* hidden = (const float*)d_in[0];
    const float* mask   = (const float*)d_in[1];
    const float* Wq = (const float*)d_in[2];   const float* bq  = (const float*)d_in[3];
    const float* Wk = (const float*)d_in[4];   const float* bk  = (const float*)d_in[5];
    const float* Wv = (const float*)d_in[6];   const float* bv  = (const float*)d_in[7];
    const float* Wo = (const float*)d_in[8];   const float* bo  = (const float*)d_in[9];
    const float* g1 = (const float*)d_in[10];  const float* b1  = (const float*)d_in[11];
    const float* Wi = (const float*)d_in[12];  const float* bi  = (const float*)d_in[13];
    const float* Wt = (const float*)d_in[14];  const float* bot = (const float*)d_in[15];
    const float* g2 = (const float*)d_in[16];  const float* b2  = (const float*)d_in[17];
    float* out = (float*)d_out;

    const size_t BSD = (size_t)BSR * DD;     // 4M
    const size_t DI  = (size_t)DD * II;      // 4M
    const size_t DDq = (size_t)DD * DD;      // 1M

    char* ws = (char*)d_ws;
    auto alloc = [&](size_t bytes) -> char* {
        char* p = ws;
        ws += (bytes + 255) & ~(size_t)255;
        return p;
    };
    unsigned short* hb    = (unsigned short*)alloc(BSD * 2);
    unsigned short* wqb   = (unsigned short*)alloc(DDq * 2);
    unsigned short* wkb   = (unsigned short*)alloc(DDq * 2);
    unsigned short* wvb   = (unsigned short*)alloc(DDq * 2);
    unsigned short* wob   = (unsigned short*)alloc(DDq * 2);
    unsigned short* wib   = (unsigned short*)alloc(DI * 2);
    unsigned short* wtb   = (unsigned short*)alloc(DI * 2);
    unsigned short* qb    = (unsigned short*)alloc(BSD * 2);
    unsigned short* kbuf  = (unsigned short*)alloc(BSD * 2);
    unsigned short* vbuf  = (unsigned short*)alloc(BSD * 2);
    unsigned short* ctxb  = (unsigned short*)alloc(BSD * 2);
    float*          y1    = (float*)alloc(BSD * 4);
    float*          attnf = (float*)alloc(BSD * 4);
    unsigned short* attnb = (unsigned short*)alloc(BSD * 2);
    unsigned short* interb= (unsigned short*)alloc((size_t)BSR * II * 2);
    float*          y2    = (float*)alloc(BSD * 4);

    const int smem_attn = 16 * 1024 * 4 + 16 * 1024 * 2;  // 96KB
    (void)hipFuncSetAttribute((const void*)attn_kernel,
                              hipFuncAttributeMaxDynamicSharedMemorySize, smem_attn);

    // 1) fp32 -> bf16 conversions
    cvt_f32_bf16<<<2048, 256, 0, stream>>>(hidden, hb, (int)BSD);
    cvt_f32_bf16<<<1024, 256, 0, stream>>>(Wq, wqb, (int)DDq);
    cvt_f32_bf16<<<1024, 256, 0, stream>>>(Wk, wkb, (int)DDq);
    cvt_f32_bf16<<<1024, 256, 0, stream>>>(Wv, wvb, (int)DDq);
    cvt_f32_bf16<<<1024, 256, 0, stream>>>(Wo, wob, (int)DDq);
    cvt_f32_bf16<<<2048, 256, 0, stream>>>(Wi, wib, (int)DI);
    cvt_f32_bf16<<<2048, 256, 0, stream>>>(Wt, wtb, (int)DI);

    // 2) QKV projections (head-scattered bf16 outputs)
    {
        dim3 g(DD / 128, BSR / 128);
        gemm_bf16<1><<<g, 256, 0, stream>>>(hb, wqb, bq, nullptr, qb,   nullptr, BSR, DD, DD);
        gemm_bf16<1><<<g, 256, 0, stream>>>(hb, wkb, bk, nullptr, kbuf, nullptr, BSR, DD, DD);
        gemm_bf16<1><<<g, 256, 0, stream>>>(hb, wvb, bv, nullptr, vbuf, nullptr, BSR, DD, DD);
    }

    // 3) attention
    attn_kernel<<<BB * NHH * (SS / 16), 32, smem_attn, stream>>>(qb, kbuf, vbuf,
                                                                 mask, ctxb);

    // 4) output projection + residual(hidden)
    {
        dim3 g(DD / 128, BSR / 128);
        gemm_bf16<0><<<g, 256, 0, stream>>>(ctxb, wob, bo, y1, nullptr, hidden,
                                            BSR, DD, DD);
    }
    // 5) LN1 -> attnf (f32) + attnb (bf16)
    ln_kernel<<<BSR, 256, 0, stream>>>(y1, g1, b1, attnf, attnb);

    // 6) FFN up + exact GELU -> bf16
    {
        dim3 g(II / 128, BSR / 128);
        gemm_bf16<2><<<g, 256, 0, stream>>>(attnb, wib, bi, nullptr, interb,
                                            nullptr, BSR, II, DD);
    }
    // 7) FFN down + residual(attnf)
    {
        dim3 g(DD / 128, BSR / 128);
        gemm_bf16<0><<<g, 256, 0, stream>>>(interb, wtb, bot, y2, nullptr, attnf,
                                            BSR, DD, II);
    }
    // 8) LN2 -> final output
    ln_kernel<<<BSR, 256, 0, stream>>>(y2, g2, b2, out, nullptr);
}
